// SE_RG_RSNN_Final_77515569758391
// MI455X (gfx1250) — matrix operand order repaired and checked by reading, not compiled, not run on hardware
//
#include <hip/hip_runtime.h>
#include <hip/hip_bf16.h>

typedef __attribute__((ext_vector_type(16))) _Float16 v16h;
typedef __attribute__((ext_vector_type(8)))  _Float16 v8h;
typedef __attribute__((ext_vector_type(8)))  float    v8f;

#define WMMA16(a,b,c) __builtin_amdgcn_wmma_f32_16x16x32_f16(false,(a),false,(b),(short)0,(c),false,false)

__device__ __forceinline__ v16h pack16(v8h lo, v8h hi){
  union { v16h v; v8h h[2]; } u; u.h[0]=lo; u.h[1]=hi; return u.v;
}

static constexpr int Bn = 128, T = 480, C1 = 128, C2 = 256;
static constexpr long BT = (long)Bn * T;       // 61440

// ---------------------------------------------------------------- weight prep
// w[k][ci][co] (f32) -> o[k][co][ci] (f16)
__global__ void k_wt(const float* __restrict__ w, _Float16* __restrict__ o,
                     int CI, int CO, long total){
  long idx = (long)blockIdx.x*blockDim.x + threadIdx.x;
  if(idx >= total) return;
  int ci = (int)(idx % CI); long r = idx / CI;
  int co = (int)(r % CO);   int k = (int)(r / CO);
  o[idx] = (_Float16)w[((long)k*CI + ci)*CO + co];
}

// ---------------------------------------------------------------- stage1 GEMM
// y1[bt][128] = x[bt][64] @ w1t^T ; w1t layout [co][ci]
__global__ __launch_bounds__(256) void k_conv1(const float* __restrict__ x,
        const _Float16* __restrict__ w1t, float* __restrict__ y1){
  __shared__ _Float16 As[16*64];
  int mt = blockIdx.x;                 // 0..3839
  int tid = threadIdx.x, lane = tid & 31, wv = tid >> 5;
  { // stage A tile (fp32 -> f16)
    int idx = tid*4; int row = idx >> 6; int col = idx & 63;
    const float* s = x + ((long)mt*16 + row)*64 + col;
    As[idx+0]=(_Float16)s[0]; As[idx+1]=(_Float16)s[1];
    As[idx+2]=(_Float16)s[2]; As[idx+3]=(_Float16)s[3];
  }
  __syncthreads();
  int l15 = lane & 15, hi = lane >> 4;
  int n0 = wv*16;
  v8f acc = {};
#pragma unroll
  for(int ch=0; ch<2; ++ch){
    int c0 = ch*32;
    v8h a0 = *(const v8h*)&As[l15*64 + c0 + hi*8];
    v8h a1 = *(const v8h*)&As[l15*64 + c0 + 16 + hi*8];
    const _Float16* bp = w1t + (n0 + l15)*64 + c0 + hi*8;
    acc = WMMA16(pack16(a0,a1), pack16(*(const v8h*)bp, *(const v8h*)(bp+16)), acc);
  }
  long base = ((long)mt*16)*C1 + n0 + l15;
#pragma unroll
  for(int j=0;j<8;++j) y1[base + (long)(hi*8+j)*C1] = acc[j];
}

// ---------------------------------------------------------------- per-(b,c) sums
__global__ void k_bt_sums(const float* __restrict__ y, float* __restrict__ S1,
                          float* __restrict__ S2, int C){
  int b = blockIdx.x, c = threadIdx.x;
  const float* p = y + (long)b*T*C + c;
  float s1=0.f, s2=0.f;
  for(int t=0;t<T;++t){ float v = p[(long)t*C]; s1 += v; s2 += v*v; }
  S1[b*C+c]=s1; S2[b*C+c]=s2;
}

// ---------------------------------------------------------------- SE MLP
template<int C, int H>
__global__ void k_se(const float* __restrict__ S1, const float* __restrict__ w1,
                     const float* __restrict__ w2, float* __restrict__ ss){
  int b = blockIdx.x, c = threadIdx.x;
  __shared__ float mean_s[C];
  __shared__ float hdn[H];
  mean_s[c] = S1[b*C+c] * (1.0f/T);
  __syncthreads();
  if(c < H){
    float a = 0.f;
    for(int i=0;i<C;++i) a += mean_s[i]*w1[i*H+c];
    hdn[c] = a > 0.f ? a : 0.f;
  }
  __syncthreads();
  float a = 0.f;
#pragma unroll
  for(int h=0;h<H;++h) a += hdn[h]*w2[h*C+c];
  ss[b*C+c] = 1.f/(1.f+__expf(-a));
}

// ---------------------------------------------------------------- BN params
template<int C>
__global__ void k_bnparams(const float* __restrict__ S1, const float* __restrict__ S2,
                           const float* __restrict__ ss, const float* __restrict__ scale,
                           const float* __restrict__ bias, float* __restrict__ g,
                           float* __restrict__ beta){
  int c = threadIdx.x;
  float sum=0.f, sum2=0.f;
  for(int b=0;b<Bn;++b){
    float s = ss[b*C+c];
    sum  += s*S1[b*C+c];
    sum2 += s*s*S2[b*C+c];
  }
  const float inv = 1.0f/((float)Bn*T);
  float m = sum*inv;
  float var = sum2*inv - m*m;
  float rs = rsqrtf(var + 1e-5f);
  float gg = scale[c]*rs;
  g[c] = gg;
  beta[c] = bias[c] - m*gg;
}

// ---------------------------------------------------------------- vth/decay
__global__ void k_lifparams(const float* __restrict__ rv, const float* __restrict__ rd,
                            const float* pvm, const float* pvs, const float* pdm,
                            const float* pds, float* __restrict__ vth,
                            float* __restrict__ dec){
  int c = threadIdx.x;
  float zv = rv[c]*pvs[0] + pvm[0];
  float sp = (zv > 20.f) ? zv : log1pf(__expf(zv));
  vth[c] = sp + 0.5f;
  float zd = rd[c]*pds[0] + pdm[0];
  dec[c] = fminf(1.f/(1.f+__expf(-zd)), 0.99f);
}

// ---------------------------------------------------------------- stage1 HLIF (no recurrence)
__global__ void k_hlif1(const float* __restrict__ y1, const float* __restrict__ ss,
                        const float* __restrict__ g, const float* __restrict__ beta,
                        const float* __restrict__ vth_, const float* __restrict__ dec_,
                        float* __restrict__ so, _Float16* __restrict__ sh){
  int idx = blockIdx.x*blockDim.x + threadIdx.x;   // b*C1 + c
  int b = idx >> 7, c = idx & 127;
  float mult = ss[idx]*g[c], add = beta[c];
  float vth = vth_[c], dec = dec_[c];
  const float* p  = y1 + (long)b*T*C1 + c;
  float*       po = so + (long)b*T*C1 + c;
  _Float16*    ph = sh + (long)b*T*C1 + c;
  float v = 0.f;
  for(int t=0;t<T;++t){
    v = v*dec + p[(long)t*C1]*mult + add;
    float s = (v - vth) > 0.f ? 1.f : 0.f;
    v -= s*vth;
    po[(long)t*C1] = s;
    ph[(long)t*C1] = (_Float16)s;
  }
}

// ---------------------------------------------------------------- dilated temporal conv (WMMA)
// 32 output rows per block (2 M-tiles/wave, B reused), double-buffered LDS staging.
// out[bt][256] = sum_k spk[b, t-PADL+DIL*k, :] @ wt[k]^T ; wt layout [k][co][ci]
template<int CIN, int DIL, int PADL>
__global__ __launch_bounds__(256) void k_dilconv(const _Float16* __restrict__ spk,
        const _Float16* __restrict__ wt, float* __restrict__ y){
  constexpr int HPT = 32*CIN/256;            // halves staged per thread (16 or 32)
  __shared__ _Float16 As[2][32*CIN];
  int mt = blockIdx.x;                        // 0..1919
  int bb = mt/15, t0 = (mt%15)*32;
  int tid = threadIdx.x, lane = tid & 31, wv = tid >> 5;
  int l15 = lane & 15, hi = lane >> 4;
  int n0a = wv*16, n0b = (wv+8)*16;
  const long rowbase = (long)bb*T;

  const int sidx = tid*HPT;
  const int srow = sidx / CIN, scol = sidx % CIN;

  v8h pre[HPT/8];
  auto stage_load = [&](int k){
    int tin = t0 + srow - PADL + DIL*k;
    if(tin >= 0 && tin < T){
      const _Float16* src = spk + (rowbase + tin)*CIN + scol;
#pragma unroll
      for(int q=0;q<HPT/8;++q) pre[q] = *(const v8h*)(src + q*8);
    } else {
      v8h z = {};
#pragma unroll
      for(int q=0;q<HPT/8;++q) pre[q] = z;
    }
  };
  auto stage_store = [&](int buf){
#pragma unroll
    for(int q=0;q<HPT/8;++q) *(v8h*)&As[buf][sidx + q*8] = pre[q];
  };

  v8f acc00={}, acc01={}, acc10={}, acc11={};   // [mtile][ntile]
  stage_load(0); stage_store(0);
  __syncthreads();
  for(int k=0;k<32;++k){
    if(k < 31) stage_load(k+1);                 // prefetch next tap (global)
    const _Float16* wk = wt + (long)k*256*CIN;
    const _Float16* A0base = &As[k&1][0];
#pragma unroll
    for(int ch=0; ch<CIN/32; ++ch){
      int c0 = ch*32;
      // A fragments for both M-tiles (rows l15 and 16+l15)
      v16h A0 = pack16(*(const v8h*)&A0base[ l15      *CIN + c0 + hi*8],
                       *(const v8h*)&A0base[ l15      *CIN + c0 + 16 + hi*8]);
      v16h A1 = pack16(*(const v8h*)&A0base[(16+l15)  *CIN + c0 + hi*8],
                       *(const v8h*)&A0base[(16+l15)  *CIN + c0 + 16 + hi*8]);
      // B fragments loaded once, reused by both M-tiles
      const _Float16* bp0 = wk + (n0a+l15)*CIN + c0 + hi*8;
      v16h B0 = pack16(*(const v8h*)bp0, *(const v8h*)(bp0+16));
      const _Float16* bp1 = wk + (n0b+l15)*CIN + c0 + hi*8;
      v16h B1 = pack16(*(const v8h*)bp1, *(const v8h*)(bp1+16));
      acc00 = WMMA16(A0, B0, acc00);
      acc01 = WMMA16(A0, B1, acc01);
      acc10 = WMMA16(A1, B0, acc10);
      acc11 = WMMA16(A1, B1, acc11);
    }
    if(k < 31) stage_store((k+1)&1);            // LDS store after compute
    __syncthreads();
  }
  long base = ((long)mt*32)*C2;
#pragma unroll
  for(int j=0;j<8;++j){
    long r0 = base + (long)(hi*8+j)*C2;
    long r1 = base + (long)(16+hi*8+j)*C2;
    y[r0 + n0a + l15] = acc00[j];
    y[r0 + n0b + l15] = acc01[j];
    y[r1 + n0a + l15] = acc10[j];
    y[r1 + n0b + l15] = acc11[j];
  }
}

// ---------------------------------------------------------------- persistent recurrent LIF/ALIF
// One block owns 16 batch rows x all 256 channels. w_rec fragments in registers,
// binary spikes in LDS (A-fragments), membrane potential in D-layout registers.
template<bool ALIF>
__global__ __launch_bounds__(256) void k_rec(const float* __restrict__ ycv,
        const _Float16* __restrict__ wrect,          // [co][ci] f16
        const float* __restrict__ ss, const float* __restrict__ g,
        const float* __restrict__ beta,
        const float* __restrict__ vth_, const float* __restrict__ dec_,   // hlif mode
        const float* __restrict__ pd,  const float* __restrict__ padp,
        const float* __restrict__ pbeta,                                   // alif mode
        float* __restrict__ so, _Float16* __restrict__ sh){
  __shared__ _Float16 prevS[16*C2];
  int tid = threadIdx.x, lane = tid & 31, wv = tid >> 5;
  int l15 = lane & 15, hi = lane >> 4;
  int b0 = blockIdx.x * 16;
  int c0idx = wv*16 + l15, c1idx = (wv+8)*16 + l15;

  // preload w_rec B-fragments (2 tiles x 8 K-chunks = 128 VGPRs)
  v16h Breg[16];
#pragma unroll
  for(int tl=0; tl<2; ++tl)
#pragma unroll
    for(int ch=0; ch<8; ++ch){
      int n0 = (wv + tl*8)*16;
      const _Float16* bp = wrect + (n0+l15)*C2 + ch*32 + hi*8;
      Breg[tl*8+ch] = pack16(*(const v8h*)bp, *(const v8h*)(bp+16));
    }

  // per-element affine and neuron params
  float mult0[8], mult1[8], v0[8]={}, v1[8]={}, th0[8]={}, th1[8]={};
#pragma unroll
  for(int j=0;j<8;++j){
    int b = b0 + hi*8 + j;
    mult0[j] = ss[b*C2 + c0idx]*g[c0idx];
    mult1[j] = ss[b*C2 + c1idx]*g[c1idx];
  }
  float add0 = beta[c0idx], add1 = beta[c1idx];
  float dec0, dec1, vthA = 0.f, vthB = 0.f, adp = 0.f, bspk = 0.f;
  if(ALIF){ dec0 = dec1 = pd[0]; adp = padp[0]; bspk = pbeta[0]; }
  else    { dec0 = dec_[c0idx]; dec1 = dec_[c1idx]; vthA = vth_[c0idx]; vthB = vth_[c1idx]; }

  for(int i = tid*16; i < tid*16+16; ++i) prevS[i] = (_Float16)0.f;
  __syncthreads();

  for(int t=0; t<T; ++t){
    // prefetch this step's conv-drive early (independent of prevS)
    float x0[8], x1[8];
#pragma unroll
    for(int j=0;j<8;++j){
      long xoff = ((long)(b0 + hi*8 + j)*T + t)*C2;
      x0[j] = ycv[xoff + c0idx];
      x1[j] = ycv[xoff + c1idx];
    }
    v8f acc0 = {}, acc1 = {};
#pragma unroll
    for(int ch=0; ch<8; ++ch){
      int c0 = ch*32;
      v8h a0 = *(const v8h*)&prevS[l15*C2 + c0 + hi*8];
      v8h a1 = *(const v8h*)&prevS[l15*C2 + c0 + 16 + hi*8];
      v16h A = pack16(a0,a1);
      acc0 = WMMA16(A, Breg[ch],   acc0);
      acc1 = WMMA16(A, Breg[8+ch], acc1);
    }
    __syncthreads();                       // all A-reads done before spike writes
#pragma unroll
    for(int j=0;j<8;++j){
      int m = hi*8 + j;
      long xoff = ((long)(b0+m)*T + t)*C2;
      // tile 0
      v0[j] = v0[j]*dec0 + x0[j]*mult0[j] + add0 + acc0[j];
      float vt = ALIF ? (0.5f + th0[j]) : vthA;
      float s = (v0[j] - vt) > 0.f ? 1.f : 0.f;
      v0[j] -= s*vt;
      if(ALIF) th0[j] = th0[j]*adp + s*bspk;
      prevS[m*C2 + c0idx] = (_Float16)s;
      so[xoff + c0idx] = s;
      if(sh) sh[xoff + c0idx] = (_Float16)s;
      // tile 1
      v1[j] = v1[j]*dec1 + x1[j]*mult1[j] + add1 + acc1[j];
      vt = ALIF ? (0.5f + th1[j]) : vthB;
      s = (v1[j] - vt) > 0.f ? 1.f : 0.f;
      v1[j] -= s*vt;
      if(ALIF) th1[j] = th1[j]*adp + s*bspk;
      prevS[m*C2 + c1idx] = (_Float16)s;
      so[xoff + c1idx] = s;
      if(sh) sh[xoff + c1idx] = (_Float16)s;
    }
    __syncthreads();                       // spikes visible before next step reads
  }
}

// ---------------------------------------------------------------- readout
__global__ __launch_bounds__(256) void k_readout(const float* __restrict__ spk3,
        const float* __restrict__ cw, const float* __restrict__ cb,
        const float* __restrict__ a1w, const float* __restrict__ a1b,
        const float* __restrict__ a2w, const float* __restrict__ a2b,
        const float* __restrict__ pdec, float* __restrict__ logits){
  int b = blockIdx.x, tid = threadIdx.x;
  __shared__ float dout[480][4];
  __shared__ float dp[20][4];
  __shared__ float attn[20];
  for(int t = tid; t < 480; t += 256){
    const float* p = spk3 + ((long)b*T + t)*C2;
    float a0=0,a1=0,a2=0,a3=0;
    for(int ci=0; ci<C2; ++ci){
      float v = p[ci];
      a0 += v*cw[ci*4+0]; a1 += v*cw[ci*4+1]; a2 += v*cw[ci*4+2]; a3 += v*cw[ci*4+3];
    }
    dout[t][0]=a0+cb[0]; dout[t][1]=a1+cb[1]; dout[t][2]=a2+cb[2]; dout[t][3]=a3+cb[3];
  }
  __syncthreads();
  if(tid < 4){
    float dec = pdec[0], v = 0.f;
    for(int gq=0; gq<20; ++gq) dp[gq][tid] = 0.f;
    for(int t=0; t<480; ++t){
      v = v*dec + dout[t][tid]*(1.f - dec);
      dp[t/24][tid] += ((t%24) >= 12 ? v : -v) * (1.f/12.f);
    }
  }
  __syncthreads();
  if(tid < 20){
    float a = a2b[0];
    for(int k=0;k<8;++k){
      float z = a1b[k];
      for(int c=0;c<4;++c) z += dp[tid][c]*a1w[c*8+k];
      a += (z>0.f? z:0.f)*a2w[k];
    }
    attn[tid] = a;
  }
  __syncthreads();
  if(tid == 0){
    float mx = -1e30f;
    for(int gq=0;gq<20;++gq) mx = fmaxf(mx, attn[gq]);
    float ssum = 0.f;
    for(int gq=0;gq<20;++gq){ attn[gq] = __expf(attn[gq]-mx); ssum += attn[gq]; }
    float inv = 1.f/ssum;
    for(int k=0;k<4;++k){
      float o = 0.f;
      for(int gq=0;gq<20;++gq) o += dp[gq][k]*attn[gq]*inv;
      logits[b*4+k] = o;
    }
  }
}

// ---------------------------------------------------------------- launch
extern "C" void kernel_launch(void* const* d_in, const int* in_sizes, int n_in,
                              void* d_out, int out_size, void* d_ws, size_t ws_size,
                              hipStream_t stream) {
  (void)in_sizes; (void)n_in; (void)out_size; (void)ws_size;
  const float* x       = (const float*)d_in[0];
  const float* conv1_w = (const float*)d_in[1];
  const float* se1_w1  = (const float*)d_in[2];
  const float* se1_w2  = (const float*)d_in[3];
  const float* bn1_s   = (const float*)d_in[4];
  const float* bn1_b   = (const float*)d_in[5];
  const float* vthr1   = (const float*)d_in[6];
  const float* decr1   = (const float*)d_in[7];
  const float* conv2_w = (const float*)d_in[8];
  const float* se2_w1  = (const float*)d_in[9];
  const float* se2_w2  = (const float*)d_in[10];
  const float* bn2_s   = (const float*)d_in[11];
  const float* bn2_b   = (const float*)d_in[12];
  const float* vthr2   = (const float*)d_in[13];
  const float* decr2   = (const float*)d_in[14];
  const float* w_rec2  = (const float*)d_in[15];
  const float* conv3_w = (const float*)d_in[16];
  const float* se3_w1  = (const float*)d_in[17];
  const float* se3_w2  = (const float*)d_in[18];
  const float* bn3_s   = (const float*)d_in[19];
  const float* bn3_b   = (const float*)d_in[20];
  const float* w_rec3  = (const float*)d_in[21];
  const float* cls_w   = (const float*)d_in[22];
  const float* cls_b   = (const float*)d_in[23];
  const float* at1w    = (const float*)d_in[24];
  const float* at1b    = (const float*)d_in[25];
  const float* at2w    = (const float*)d_in[26];
  const float* at2b    = (const float*)d_in[27];
  const float* v1m=(const float*)d_in[28], *v1s=(const float*)d_in[29];
  const float* d1m=(const float*)d_in[30], *d1s=(const float*)d_in[31];
  const float* v2m=(const float*)d_in[33], *v2s=(const float*)d_in[34];
  const float* d2m=(const float*)d_in[35], *d2s=(const float*)d_in[36];
  const float* alif_d=(const float*)d_in[38], *alif_adp=(const float*)d_in[39];
  const float* alif_beta=(const float*)d_in[40];
  const float* dec_out=(const float*)d_in[42];

  // workspace carve-up
  size_t off = 0;
  auto carve = [&](size_t bytes)->char*{
    char* p = (char*)d_ws + off;
    off = (off + bytes + 255) & ~(size_t)255;
    return p;
  };
  _Float16* w1t  = (_Float16*)carve(128*64*2);
  _Float16* w2t  = (_Float16*)carve((size_t)32*256*128*2);
  _Float16* w3t  = (_Float16*)carve((size_t)32*256*256*2);
  _Float16* wr2t = (_Float16*)carve(256*256*2);
  _Float16* wr3t = (_Float16*)carve(256*256*2);
  float* y1 = (float*)carve(BT*C1*4);
  float* y2 = (float*)carve(BT*C2*4);
  float* y3 = (float*)carve(BT*C2*4);
  _Float16* s1h = (_Float16*)carve(BT*C1*2);
  _Float16* s2h = (_Float16*)carve(BT*C2*2);
  float* S1  = (float*)carve(Bn*C2*4);
  float* S2  = (float*)carve(Bn*C2*4);
  float* ssc = (float*)carve(Bn*C2*4);
  float* gch  = (float*)carve(C2*4);
  float* bch  = (float*)carve(C2*4);
  float* vthc = (float*)carve(C2*4);
  float* decc = (float*)carve(C2*4);

  float* out    = (float*)d_out;
  float* logits = out;
  float* spk1o  = out + 512;
  float* spk2o  = spk1o + BT*C1;
  float* spk3o  = spk2o + BT*C2;

  // weight prep
  k_wt<<<(128*64+255)/256, 256, 0, stream>>>(conv1_w, w1t, 64, 128, 128*64);
  k_wt<<<((long)32*256*128+255)/256, 256, 0, stream>>>(conv2_w, w2t, 128, 256, (long)32*256*128);
  k_wt<<<((long)32*256*256+255)/256, 256, 0, stream>>>(conv3_w, w3t, 256, 256, (long)32*256*256);
  k_wt<<<(256*256+255)/256, 256, 0, stream>>>(w_rec2, wr2t, 256, 256, 256*256);
  k_wt<<<(256*256+255)/256, 256, 0, stream>>>(w_rec3, wr3t, 256, 256, 256*256);

  // stage 1
  k_conv1<<<3840, 256, 0, stream>>>(x, w1t, y1);
  k_bt_sums<<<Bn, C1, 0, stream>>>(y1, S1, S2, C1);
  k_se<128,8><<<Bn, 128, 0, stream>>>(S1, se1_w1, se1_w2, ssc);
  k_bnparams<128><<<1, 128, 0, stream>>>(S1, S2, ssc, bn1_s, bn1_b, gch, bch);
  k_lifparams<<<1, C1, 0, stream>>>(vthr1, decr1, v1m, v1s, d1m, d1s, vthc, decc);
  k_hlif1<<<(Bn*C1)/256, 256, 0, stream>>>(y1, ssc, gch, bch, vthc, decc, spk1o, s1h);

  // stage 2
  k_dilconv<128,4,62><<<1920, 256, 0, stream>>>(s1h, w2t, y2);
  k_bt_sums<<<Bn, C2, 0, stream>>>(y2, S1, S2, C2);
  k_se<256,16><<<Bn, 256, 0, stream>>>(S1, se2_w1, se2_w2, ssc);
  k_bnparams<256><<<1, 256, 0, stream>>>(S1, S2, ssc, bn2_s, bn2_b, gch, bch);
  k_lifparams<<<1, C2, 0, stream>>>(vthr2, decr2, v2m, v2s, d2m, d2s, vthc, decc);
  k_rec<false><<<Bn/16, 256, 0, stream>>>(y2, wr2t, ssc, gch, bch, vthc, decc,
                                          nullptr, nullptr, nullptr, spk2o, s2h);

  // stage 3
  k_dilconv<256,12,186><<<1920, 256, 0, stream>>>(s2h, w3t, y3);
  k_bt_sums<<<Bn, C2, 0, stream>>>(y3, S1, S2, C2);
  k_se<256,16><<<Bn, 256, 0, stream>>>(S1, se3_w1, se3_w2, ssc);
  k_bnparams<256><<<1, 256, 0, stream>>>(S1, S2, ssc, bn3_s, bn3_b, gch, bch);
  k_rec<true><<<Bn/16, 256, 0, stream>>>(y3, wr3t, ssc, gch, bch, nullptr, nullptr,
                                         alif_d, alif_adp, alif_beta, spk3o, nullptr);

  // readout
  k_readout<<<Bn, 256, 0, stream>>>(spk3o, cls_w, cls_b, at1w, at1b, at2w, at2b,
                                    dec_out, logits);
}
